// MODS_40501541601760
// MI455X (gfx1250) — compile-verified, matrix operand order checked by reading, not run on hardware
//
#include <hip/hip_runtime.h>
#include <hip/hip_bf16.h>

typedef __attribute__((ext_vector_type(16))) _Float16 v16h;
typedef __attribute__((ext_vector_type(8)))  float    v8f;

#define LLEN 2048
#define MORD 21
#define WIN  10
#define NW   8
#define NT   256
#define LTILE 16
#define NTILES (LLEN / LTILE)
#define NFRAG 7   // 3 x w1^T N-tiles, 4 x w2^T (2 N-tiles x 2 K-chunks)

// K index held by (element j, lane-half hi) in the 16-bit 16x32 A/B fragment layout
// (ISA 7.12.2): VGPR v<4: lanes0-15 K=2v..2v+1, lanes16-31 K=8+2v..; v>=4: +16/+24.
__device__ __forceinline__ int kmap(int j, int hi) {
    int v = j >> 1, lo = j & 1;
    int base = (v < 4) ? (hi ? 8 : 0) : (hi ? 16 : 8);
    return 2 * v + lo + base;
}

__device__ __forceinline__ float gelu_exact(float v) {
    return 0.5f * v * (1.0f + erff(v * 0.70710678118654752f));
}

__global__ __launch_bounds__(NT, 4) void MODS_fracdiff_mlp_kernel(
    const float* __restrict__ x, const float* __restrict__ orders,
    const float* __restrict__ ln_w, const float* __restrict__ ln_b,
    const float* __restrict__ w1, const float* __restrict__ b1,
    const float* __restrict__ w2, const float* __restrict__ b2,
    float* __restrict__ out)
{
    __shared__ float xs[LLEN + WIN];          // xs[0..8]=0 pad; xs[9+i]=x[i]
    __shared__ float Wl[MORD * WIN];          // fracdiff coefficients
    __shared__ __align__(32) _Float16 wfrag[NFRAG][32][16];  // weight B-fragments
    __shared__ _Float16 gbuf[NW][16 * 64];    // per-wave GELU bounce (A2 staging)
    __shared__ float    sbuf[NW][16 * 32];    // per-wave D2 bounce
    __shared__ float pS[NW], pQ[NW];
    __shared__ float s_mu, s_rsig;

    const int tid  = threadIdx.x;
    const int lane = tid & 31;
    const int wv   = tid >> 5;
    const int hi   = lane >> 4;
    const int lr   = lane & 15;
    const int bc   = blockIdx.x;
    const float* xrow = x + (size_t)bc * LLEN;

    // ---- stage x (front zero-pad), fracdiff coefs ----
    if (tid < WIN - 1) xs[tid] = 0.0f;
    for (int i = tid; i < LLEN; i += NT) xs[WIN - 1 + i] = xrow[i];
    if (tid < MORD) {
        float n = orders[tid];
        float c = 1.0f;
        Wl[tid * WIN + 0] = 1.0f;
        #pragma unroll
        for (int k = 1; k < WIN; ++k) {
            c *= ((float)k - 1.0f - n) / (float)k;
            Wl[tid * WIN + k] = c;
        }
    }
    // ---- weight B-fragments (B = W^T, zero-padded), built once per block ----
    for (int idx = tid; idx < NFRAG * 32 * 16; idx += NT) {
        int f  = idx >> 9;            // fragment id
        int r  = idx & 511;
        int ln = r >> 4;
        int j  = r & 15;
        int fhi = ln >> 4, flr = ln & 15;
        float v = 0.0f;
        if (f < 3) {                          // w1^T : K(=m) pad 21->32, N-tile f
            int m = kmap(j, fhi), h = 16 * f + flr;
            if (m < MORD && h < 2 * MORD) v = w1[h * MORD + m];
        } else {                              // w2^T : K(=h) 2 chunks, N-tile t
            int t = (f - 3) >> 1, c = (f - 3) & 1;
            int h = 32 * c + kmap(j, fhi);
            int mo = 16 * t + flr;
            if (h < 2 * MORD && mo < MORD) v = w2[mo * 2 * MORD + h];
        }
        wfrag[f][ln][j] = (_Float16)v;
    }
    // zero gbuf K-pad region h=48..63 (per wave, never rewritten)
    for (int idx = lane; idx < 16 * 16; idx += 32) {
        int row = idx >> 4, col = 48 + (idx & 15);
        gbuf[wv][row * 64 + col] = (_Float16)0.0f;
    }

    float b1v[3];
    #pragma unroll
    for (int t = 0; t < 3; ++t) { int h = 16 * t + lr; b1v[t] = (h < 2 * MORD) ? b1[h] : 0.0f; }
    float sumb2 = 0.0f;
    for (int m = 0; m < MORD; ++m) sumb2 += b2[m];

    __syncthreads();

    // ---- pass 1: LayerNorm stats over the whole (L, M) slab ----
    float sum = 0.0f, sq = 0.0f;
    for (int l = tid; l < LLEN; l += NT) {
        float xw[WIN];
        #pragma unroll
        for (int k = 0; k < WIN; ++k) xw[k] = xs[WIN - 1 + l - k];
        for (int o = 0; o < MORD; ++o) {
            float d = 0.0f;
            #pragma unroll
            for (int k = 0; k < WIN; ++k) d = fmaf(Wl[o * WIN + k], xw[k], d);
            sum += d; sq += d * d;
        }
    }
    #pragma unroll
    for (int off = 16; off > 0; off >>= 1) {
        sum += __shfl_down(sum, off);
        sq  += __shfl_down(sq,  off);
    }
    if (lane == 0) { pS[wv] = sum; pQ[wv] = sq; }
    __syncthreads();
    if (tid == 0) {
        float S = 0.0f, Q = 0.0f;
        #pragma unroll
        for (int i = 0; i < NW; ++i) { S += pS[i]; Q += pQ[i]; }
        float inv = 1.0f / ((float)LLEN * (float)MORD);
        float muv = S * inv;
        float var = Q * inv - muv * muv;
        s_mu = muv; s_rsig = rsqrtf(var + 1e-6f);
    }
    __syncthreads();
    const float mu = s_mu, rsig = s_rsig;

    // ---- pass 2: per-wave 16-token tiles through WMMA MLP ----
    for (int tile = wv; tile < NTILES; tile += NW) {
        const int l0 = tile * LTILE;
        const int l  = l0 + lr;                 // this lane's token (row)
        float xw[WIN];
        #pragma unroll
        for (int k = 0; k < WIN; ++k) xw[k] = xs[WIN - 1 + l - k];

        // A fragment: normalized diffs, K pad 21->32
        v16h a;
        #pragma unroll
        for (int j = 0; j < 16; ++j) {
            int o = kmap(j, hi);
            float v = 0.0f;
            if (o < MORD) {
                float d = 0.0f;
                #pragma unroll
                for (int k = 0; k < WIN; ++k) d = fmaf(Wl[o * WIN + k], xw[k], d);
                v = (d - mu) * rsig * ln_w[l * MORD + o] + ln_b[l * MORD + o];
            }
            a[j] = (_Float16)v;
        }

        // GEMM1: (16x32) x (32x16) x3 N-tiles -> f32 (B fragments from LDS)
        v8f zc = {};
        v8f d1[3];
        #pragma unroll
        for (int t = 0; t < 3; ++t) {
            v16h bf = *(const v16h*)&wfrag[t][lane][0];
            d1[t] = __builtin_amdgcn_wmma_f32_16x16x32_f16(
                        false, a, false, bf, (short)0, zc, false, false);
        }

        // bias + exact GELU -> LDS bounce in row-major (A2 staging)
        #pragma unroll
        for (int t = 0; t < 3; ++t) {
            int h = 16 * t + lr;
            bool act = (h < 2 * MORD);
            #pragma unroll
            for (int i = 0; i < 8; ++i) {
                int row = i + 8 * hi;
                float v = d1[t][i] + b1v[t];
                float g = act ? gelu_exact(v) : 0.0f;
                gbuf[wv][row * 64 + h] = (_Float16)g;
            }
        }

        // repack as A2 fragments (K = 42 padded to 64, 2 chunks)
        v16h a2c0, a2c1;
        #pragma unroll
        for (int j = 0; j < 16; ++j) {
            int kk = kmap(j, hi);
            a2c0[j] = gbuf[wv][lr * 64 + kk];
            a2c1[j] = gbuf[wv][lr * 64 + 32 + kk];
        }

        // GEMM2: chained K-accumulation, 2 N-tiles -> D2 bounce
        #pragma unroll
        for (int t = 0; t < 2; ++t) {
            v16h bta = *(const v16h*)&wfrag[3 + 2 * t][lane][0];
            v16h btb = *(const v16h*)&wfrag[4 + 2 * t][lane][0];
            v8f acc = {};
            acc = __builtin_amdgcn_wmma_f32_16x16x32_f16(
                      false, a2c0, false, bta, (short)0, acc, false, false);
            acc = __builtin_amdgcn_wmma_f32_16x16x32_f16(
                      false, a2c1, false, btb, (short)0, acc, false, false);
            #pragma unroll
            for (int i = 0; i < 8; ++i) {
                int row = i + 8 * hi;
                sbuf[wv][row * 32 + 16 * t + lr] = acc[i];
            }
        }

        // channel mean + residual (16 lanes, one token each)
        if (lane < 16) {
            float s = 0.0f;
            #pragma unroll
            for (int m = 0; m < MORD; ++m) s += sbuf[wv][lane * 32 + m];
            out[(size_t)bc * LLEN + l0 + lane] =
                xs[WIN - 1 + l0 + lane] + (s + sumb2) * (1.0f / (float)MORD);
        }
    }
}

extern "C" void kernel_launch(void* const* d_in, const int* in_sizes, int n_in,
                              void* d_out, int out_size, void* d_ws, size_t ws_size,
                              hipStream_t stream) {
    const float* x      = (const float*)d_in[0];
    const float* orders = (const float*)d_in[1];
    const float* ln_w   = (const float*)d_in[2];
    const float* ln_b   = (const float*)d_in[3];
    const float* w1     = (const float*)d_in[4];
    const float* b1     = (const float*)d_in[5];
    const float* w2     = (const float*)d_in[6];
    const float* b2     = (const float*)d_in[7];
    float* out = (float*)d_out;

    const int BC = in_sizes[0] / LLEN;   // 64*23 = 1472 rows
    MODS_fracdiff_mlp_kernel<<<BC, NT, 0, stream>>>(
        x, orders, ln_w, ln_b, w1, b1, w2, b2, out);
}
